// TDRumorGCN_22643067584840
// MI455X (gfx1250) — matrix operand order repaired and checked by reading, not compiled, not run on hardware
//
#include <hip/hip_runtime.h>

// Problem constants (match reference).
#define N_NODES   40000
#define IN_FEATS  5000
#define N_EDGES   80000
#define N_GRAPHS  128

typedef __attribute__((ext_vector_type(16))) __bf16 v16bf;
typedef __attribute__((ext_vector_type(8)))  float  v8f;

__device__ __forceinline__ unsigned pack_bf16(float a, float b) {
  unsigned ua = __builtin_bit_cast(unsigned, a);
  unsigned ub = __builtin_bit_cast(unsigned, b);
  // round-to-nearest-even fp32 -> bf16
  ua = (ua + 0x7fffu + ((ua >> 16) & 1u)) >> 16;
  ub = (ub + 0x7fffu + ((ub >> 16) & 1u)) >> 16;
  return (ua & 0xffffu) | (ub << 16);
}

// 64B global -> LDS async copy (4 x b128), tracked by ASYNCcnt.
__device__ __forceinline__ void async_copy64B(unsigned lds_off, const float* g) {
#pragma unroll
  for (int i = 0; i < 4; ++i) {
    unsigned long long ga = (unsigned long long)(const void*)(g + 4 * i);
    asm volatile("global_load_async_to_lds_b128 %0, %1, off"
                 :: "v"(lds_off + 16u * i), "v"(ga) : "memory");
  }
}

// ---------------------------------------------------------------------------
// Tiled GEMM: Out[M x 64] = op(A)[M x K] @ W[K x 64]
//   A: fp32, row stride lda; optional row gather (rowIdx) + relu (on A).
//   Wp: weights pre-packed as bf16 pairs, layout [K/2][64] u32 (K even).
//   128 rows/block, 8 wave32 waves x (16 rows x 64 cols), K-step 32.
//   A tiles double-buffered in LDS via async global->LDS copies; bf16
//   conversion happens at fragment build. v_wmma_f32_16x16x32_bf16 accum.
// ---------------------------------------------------------------------------
__global__ __launch_bounds__(256)
void k_gemm_bf16(const float* __restrict__ A, const long long* __restrict__ rowIdx,
                 int applyRelu, const unsigned* __restrict__ Wp,
                 float* __restrict__ Out, int M, int K, int lda)
{
  __shared__ float    lAf[2][128 * 32];  // raw fp32 A tiles, 2 x 16KB
  __shared__ unsigned lBu[2][64 * 16];   // packed bf16-pair B tiles, 2 x 4KB

  const int t    = threadIdx.x;
  const int lane = t & 31;
  const int wave = t >> 5;
  const int hi   = lane >> 4;
  const int l15  = lane & 15;
  const int row0 = blockIdx.x * 128;

  // A loader: 2 threads/row, 16 contiguous floats (64B) each.
  const int a_row  = t >> 1;
  const int a_half = t & 1;
  int src_row = row0 + a_row;
  if (src_row > M - 1) src_row = M - 1;          // clamp: dropped at store
  const long long g = rowIdx ? rowIdx[src_row] : (long long)src_row;
  const float* a_ptr = A + g * (long long)lda + a_half * 16;

  unsigned lds_a[2];
  lds_a[0] = (unsigned)(unsigned long long)(void*)&lAf[0][a_row * 32 + a_half * 16];
  lds_a[1] = (unsigned)(unsigned long long)(void*)&lAf[1][a_row * 32 + a_half * 16];

  // B loader: n = t&63, 4 pair-slots per thread.
  const int b_n  = t & 63;
  const int b_s0 = t >> 6;
  const int Kp   = K >> 1;

  v8f acc[4] = {v8f{}, v8f{}, v8f{}, v8f{}};
  const int mloc = wave * 16 + l15;

  // A fragment K indices (ISA 16-bit A 16x32 layout), B slots in the same k order.
  int kfrag[8];
#pragma unroll
  for (int p = 0; p < 8; ++p)
    kfrag[p] = ((p < 4) ? 2 * p : 2 * (p - 4) + 16) + 8 * hi;

  auto do_tile = [&](const float* As, const unsigned* Bs) {
    union { v16bf v; unsigned u[8]; } af;
#pragma unroll
    for (int p = 0; p < 8; ++p) {
      const float2 w = *(const float2*)&As[mloc * 32 + kfrag[p]];
      float f0 = w.x, f1 = w.y;
      if (applyRelu) { f0 = fmaxf(f0, 0.0f); f1 = fmaxf(f1, 0.0f); }
      af.u[p] = pack_bf16(f0, f1);
    }
#pragma unroll
    for (int nt = 0; nt < 4; ++nt) {
      union { v16bf v; unsigned u[8]; } bf;
      const int n = nt * 16 + l15;
#pragma unroll
      for (int p = 0; p < 8; ++p) bf.u[p] = Bs[n * 16 + ((p < 4) ? p : p + 4) + 4 * hi];
      acc[nt] = __builtin_amdgcn_wmma_f32_16x16x32_bf16(
          false, af.v, false, bf.v, (short)0, acc[nt], false, false);
    }
  };

  const int nk    = K >> 5;   // full 32-wide K steps
  const int ktail = K & 31;

  if (nk > 0) async_copy64B(lds_a[0], a_ptr);   // prologue prefetch

  for (int kt = 0; kt < nk; ++kt) {
    const int cur = kt & 1;
    const int k0  = kt << 5;
    // stage B tile (raw packed u32 copy, unguarded: full step)
#pragma unroll
    for (int s = 0; s < 4; ++s) {
      const int slot = b_s0 + s * 4;
      lBu[cur][b_n * 16 + slot] = Wp[(long long)((k0 >> 1) + slot) * 64 + b_n];
    }
    // prefetch next A tile while waiting for current
    if (kt + 1 < nk) {
      async_copy64B(lds_a[(kt + 1) & 1], a_ptr + (kt + 1) * 32);
      asm volatile("s_wait_asynccnt 0x4" ::: "memory");  // current 4 done
    } else {
      asm volatile("s_wait_asynccnt 0x0" ::: "memory");
    }
    __syncthreads();
    do_tile(lAf[cur], lBu[cur]);
    __syncthreads();   // protect buffers from next iteration's writers
  }

  if (ktail) {  // guarded tail step (runs at most once; K=5000 -> 8 cols)
    const int cur = nk & 1;
    const int k0  = nk << 5;
#pragma unroll
    for (int j = 0; j < 16; ++j) {
      const int k = k0 + a_half * 16 + j;
      lAf[cur][a_row * 32 + a_half * 16 + j] = (k < K) ? a_ptr[k0 + j] : 0.0f;
    }
#pragma unroll
    for (int s = 0; s < 4; ++s) {
      const int slot = b_s0 + s * 4;
      const int kp   = (k0 >> 1) + slot;
      lBu[cur][b_n * 16 + slot] = (kp < Kp) ? Wp[(long long)kp * 64 + b_n] : 0u;
    }
    __syncthreads();
    do_tile(lAf[cur], lBu[cur]);
  }

  // store (C/D layout: m = q + 8*hi, n = nt*16 + (lane&15))
#pragma unroll
  for (int nt = 0; nt < 4; ++nt) {
    const int n = nt * 16 + l15;
#pragma unroll
    for (int q = 0; q < 8; ++q) {
      const int m = row0 + wave * 16 + q + 8 * hi;
      if (m < M) Out[(long long)m * 64 + n] = acc[nt][q];
    }
  }
}

// ---------------------------------------------------------------------------
// Weight pre-pack: W [K x 64] fp32 -> Wp [K/2 x 64] u32 of bf16 pairs.
// ---------------------------------------------------------------------------
__global__ void k_packW(const float* __restrict__ W, unsigned* __restrict__ Wp, int Kp) {
  int i = blockIdx.x * blockDim.x + threadIdx.x;
  if (i < Kp * 64) {
    int kp = i >> 6, n = i & 63;
    Wp[i] = pack_bf16(W[(long long)(2 * kp) * 64 + n],
                      W[(long long)(2 * kp + 1) * 64 + n]);
  }
}

// ---------------------------------------------------------------------------
// Elementwise / scatter kernels
// ---------------------------------------------------------------------------
__global__ void k_set_deg(float* deg) {
  int i = blockIdx.x * blockDim.x + threadIdx.x;
  if (i < N_NODES) deg[i] = 1.0f;  // self loop
}

__global__ void k_deg_edges(const long long* __restrict__ ei, float* deg) {
  int e = blockIdx.x * blockDim.x + threadIdx.x;
  if (e < N_EDGES) unsafeAtomicAdd(&deg[ei[N_EDGES + e]], 1.0f);
}

__global__ void k_dinv(float* deg) {
  int i = blockIdx.x * blockDim.x + threadIdx.x;
  if (i < N_NODES) deg[i] = rsqrtf(deg[i]);  // deg >= 1 always
}

// agg[v][f] = bias[f] + dinv[v]^2 * hin[v][f]
__global__ void k_agg_init(const float* __restrict__ hin, const float* __restrict__ dinv,
                           const float* __restrict__ bias, float* __restrict__ agg) {
  int i = blockIdx.x * blockDim.x + threadIdx.x;
  if (i < N_NODES * 64) {
    int v = i >> 6, f = i & 63;
    float d = dinv[v];
    agg[i] = bias[f] + d * d * hin[i];
  }
}

// agg[col][f] += dinv[row]*dinv[col] * hin[row][f]
__global__ void k_agg_edges(const float* __restrict__ hin, const float* __restrict__ dinv,
                            const long long* __restrict__ ei, float* __restrict__ agg) {
  int i = blockIdx.x * blockDim.x + threadIdx.x;
  if (i < N_EDGES * 64) {
    int e = i >> 6, f = i & 63;
    long long r = ei[e], c = ei[N_EDGES + e];
    unsafeAtomicAdd(&agg[c * 64 + f], dinv[r] * dinv[c] * hin[r * 64 + f]);
  }
}

// h[v][f] += R[batch[v]][f]
__global__ void k_add_root(float* __restrict__ h, const float* __restrict__ R,
                           const long long* __restrict__ batch) {
  int i = blockIdx.x * blockDim.x + threadIdx.x;
  if (i < N_NODES * 64) {
    int v = i >> 6, f = i & 63;
    h[i] += R[batch[v] * 64 + f];
  }
}

__global__ void k_zero(float* out, float* counts) {
  int i = blockIdx.x * blockDim.x + threadIdx.x;
  if (i < N_GRAPHS * 128) out[i] = 0.0f;
  if (i < N_GRAPHS) counts[i] = 0.0f;
}

__global__ void k_pool_accum(const float* __restrict__ agg2, const long long* __restrict__ batch,
                             float* __restrict__ out, float* __restrict__ counts) {
  int i = blockIdx.x * blockDim.x + threadIdx.x;
  if (i < N_NODES * 64) {
    int v = i >> 6, f = i & 63;
    long long b = batch[v];
    unsafeAtomicAdd(&out[b * 128 + f], fmaxf(agg2[i], 0.0f));
    if (f == 0) unsafeAtomicAdd(&counts[b], 1.0f);
  }
}

__global__ void k_pool_final(float* __restrict__ out, const float* __restrict__ counts,
                             const float* __restrict__ agg1, const long long* __restrict__ root) {
  int i = blockIdx.x * blockDim.x + threadIdx.x;
  if (i < N_GRAPHS * 128) {
    int b = i >> 7, f = i & 127;
    float c = counts[b];
    float d = fmaxf(c, 1.0f);
    if (f < 64) out[i] = out[i] / d;
    else        out[i] = (c / d) * agg1[root[b] * 64 + (f - 64)];
  }
}

// ---------------------------------------------------------------------------
extern "C" void kernel_launch(void* const* d_in, const int* in_sizes, int n_in,
                              void* d_out, int out_size, void* d_ws, size_t ws_size,
                              hipStream_t stream) {
  const float*     x     = (const float*)d_in[0];
  const float*     W1    = (const float*)d_in[1];
  const float*     b1    = (const float*)d_in[2];
  const float*     W2    = (const float*)d_in[3];   // [(64+5000) x 64]
  const float*     b2    = (const float*)d_in[4];
  const long long* ei    = (const long long*)d_in[5];
  const long long* batch = (const long long*)d_in[6];
  const long long* root  = (const long long*)d_in[7];
  float*           out   = (float*)d_out;

  // Workspace: dinv | h1 | agg1 | h2 | R | counts | Wp1 | WpBot | WpTop
  float* dinv   = (float*)d_ws;
  float* h1     = dinv + N_NODES;
  float* agg1   = h1 + (size_t)N_NODES * 64;
  float* h2     = agg1 + (size_t)N_NODES * 64;
  float* R      = h2 + (size_t)N_NODES * 64;
  float* counts = R + N_GRAPHS * 64;
  unsigned* Wp1   = (unsigned*)(counts + N_GRAPHS);
  unsigned* WpBot = Wp1 + (size_t)(IN_FEATS / 2) * 64;
  unsigned* WpTop = WpBot + (size_t)(IN_FEATS / 2) * 64;
  float* agg2 = h1;  // alias: h1 dead once agg1 is built

  const int T = 256;
  const int gN    = (N_NODES + T - 1) / T;
  const int gE    = (N_EDGES + T - 1) / T;
  const int gN64  = (N_NODES * 64 + T - 1) / T;
  const int gE64  = (N_EDGES * 64 + T - 1) / T;
  const int gOut  = (N_GRAPHS * 128 + T - 1) / T;
  const int gGemm = (N_NODES + 127) / 128;
  const int gW    = ((IN_FEATS / 2) * 64 + T - 1) / T;

  // Pre-pack weights to bf16 pairs (once per call; L2-resident afterwards)
  k_packW<<<gW, T, 0, stream>>>(W1, Wp1, IN_FEATS / 2);
  k_packW<<<gW, T, 0, stream>>>(W2 + 64 * 64, WpBot, IN_FEATS / 2);
  k_packW<<<(32 * 64 + T - 1) / T, T, 0, stream>>>(W2, WpTop, 32);

  // Degree / normalization
  k_set_deg  <<<gN, T, 0, stream>>>(dinv);
  k_deg_edges<<<gE, T, 0, stream>>>(ei, dinv);
  k_dinv     <<<gN, T, 0, stream>>>(dinv);

  // conv1 linear: h1 = x @ W1  (HBM-bound: one pass over 800MB of x)
  k_gemm_bf16<<<gGemm, T, 0, stream>>>(x, nullptr, 0, Wp1, h1, N_NODES, IN_FEATS, IN_FEATS);

  // Per-graph root contribution: R = relu(x[root]) @ W2[64:,:]
  k_gemm_bf16<<<1, T, 0, stream>>>(x, root, 1, WpBot, R, N_GRAPHS, IN_FEATS, IN_FEATS);

  // conv1 aggregation (+b1)
  k_agg_init <<<gN64, T, 0, stream>>>(h1, dinv, b1, agg1);
  k_agg_edges<<<gE64, T, 0, stream>>>(h1, dinv, ei, agg1);

  // conv2 linear: h2 = relu(agg1) @ W2[:64,:] + R[batch]
  k_gemm_bf16<<<gGemm, T, 0, stream>>>(agg1, nullptr, 1, WpTop, h2, N_NODES, 64, 64);
  k_add_root <<<gN64, T, 0, stream>>>(h2, R, batch);

  // conv2 aggregation (+b2)
  k_agg_init <<<gN64, T, 0, stream>>>(h2, dinv, b2, agg2);
  k_agg_edges<<<gE64, T, 0, stream>>>(h2, dinv, ei, agg2);

  // Pooling: mean of concat(relu(agg2), agg1[root]) per graph
  k_zero      <<<gOut, T, 0, stream>>>(out, counts);
  k_pool_accum<<<gN64, T, 0, stream>>>(agg2, batch, out, counts);
  k_pool_final<<<gOut, T, 0, stream>>>(out, counts, agg1, root);
}
